// gMLPBlock_35897336660175
// MI455X (gfx1250) — compile-verified, matrix operand order checked by reading, not run on hardware
//
#include <hip/hip_runtime.h>
#include <hip/hip_bf16.h>
#include <math.h>

#define HID 128
#define FFN 256

typedef __attribute__((ext_vector_type(16))) __bf16 v16bf;
typedef __attribute__((ext_vector_type(8)))  float  v8f;
typedef __attribute__((ext_vector_type(4)))  int    v4i;

#if defined(__has_builtin)
#if __has_builtin(__builtin_amdgcn_global_load_async_to_lds_b128)
#define HAS_ASYNC_LDS 1
#endif
#endif
#ifndef HAS_ASYNC_LDS
#define HAS_ASYNC_LDS 0
#endif

typedef __attribute__((address_space(1))) v4i gas_v4i;   // global-AS 16B vector
typedef __attribute__((address_space(3))) v4i las_v4i;   // LDS-AS 16B vector

__device__ __forceinline__ unsigned short f2bf(float f) {
    unsigned u = __float_as_uint(f);
    u += 0x7FFFu + ((u >> 16) & 1u);           // round-to-nearest-even
    return (unsigned short)(u >> 16);
}

__device__ __forceinline__ float wave_sum(float v) {
    #pragma unroll
    for (int off = 16; off > 0; off >>= 1) v += __shfl_xor(v, off, 32);
    return v;
}

__device__ __forceinline__ void gatomic_add(float* p, float v) {
    // native global_atomic_add_f32 (non-returning, STOREcnt path)
    unsafeAtomicAdd(p, v);
}

// ---------------- small utility kernels ----------------
__global__ void cast_f32_bf16(const float* __restrict__ in,
                              unsigned short* __restrict__ out, int n) {
    int i = blockIdx.x * blockDim.x + threadIdx.x;
    if (i < n) out[i] = f2bf(in[i]);
}

__global__ void fill_f32(float* __restrict__ p, float v, size_t n) {
    size_t i = (size_t)blockIdx.x * blockDim.x + threadIdx.x;
    if (i < n) p[i] = v;
}

__global__ void deg_accum(const int* __restrict__ dst,
                          float* __restrict__ deg, int E) {
    int e = blockIdx.x * blockDim.x + threadIdx.x;
    if (e < E) gatomic_add(&deg[dst[e]], 1.0f);
}

__global__ void dinv_kernel(const float* __restrict__ deg,
                            float* __restrict__ dinv, int n) {
    int i = blockIdx.x * blockDim.x + threadIdx.x;
    if (i < n) {
        float d = deg[i];                       // includes 2.0 self-loop weight
        dinv[i] = d > 0.0f ? rsqrtf(d) : 0.0f;
    }
}

// ---------------- LayerNorm: one wave (32 lanes) per row -> bf16 out --------
template <int D>
__global__ void ln_rows(const float* __restrict__ x,
                        const float* __restrict__ g,
                        const float* __restrict__ b,
                        unsigned short* __restrict__ out, int M) {
    const int lane = threadIdx.x & 31;
    const int wave = threadIdx.x >> 5;
    const int row  = blockIdx.x * 8 + wave;
    if (row >= M) return;
    constexpr int PER = D / 32;
    float v[PER];
    float s = 0.0f, ss = 0.0f;
    const float* xr = x + (size_t)row * D;
    #pragma unroll
    for (int i = 0; i < PER; ++i) {
        v[i] = xr[lane + i * 32];
        s  += v[i];
        ss += v[i] * v[i];
    }
    s  = wave_sum(s);
    ss = wave_sum(ss);
    const float mu  = s / (float)D;
    const float var = ss / (float)D - mu * mu;
    const float rs  = rsqrtf(var + 1e-5f);
    unsigned short* orow = out + (size_t)row * D;
    #pragma unroll
    for (int i = 0; i < PER; ++i) {
        int c = lane + i * 32;
        orow[c] = f2bf((v[i] - mu) * rs * g[c] + b[c]);
    }
}

// ---------------- WMMA GEMM: C[M,Nc] = A[M,K](bf16) * W[Nc,K]^T + bias ------
// EPI: 0 = bias, 1 = bias + exact GELU, 2 = no bias
// Block = 8 waves; all waves share one 16-row A tile (tiles_n is 8 or 16 and
// grids are 8-tile aligned, so blockIdx*8 .. +7 never cross an M-tile row).
// A tile (16 x K bf16, <= 8KB) is staged into LDS once per block via CDNA5
// async global->LDS copies (ASYNCcnt), then each wave feeds its WMMA A
// operand from LDS (ds_load_b128) and streams its own B tile from the small
// L2-resident weight matrix.
// Operand packing follows the CDNA5 16-bit layout:
//   lanes 0-15: row = lane,    VGPR0-3 -> K {0..7},  VGPR4-7 -> K {16..23}
//   lanes16-31: row = lane-16, VGPR0-3 -> K {8..15}, VGPR4-7 -> K {24..31}
template <int EPI>
__global__ void gemm_wmma_bf16(const unsigned int* __restrict__ A32,  // bf16 [M,K]
                               const unsigned int* __restrict__ W32,  // bf16 [Nc,K]
                               const float* __restrict__ bias,
                               float* __restrict__ C,
                               int M, int K, int Nc) {
    __shared__ alignas(16) unsigned short shA[16 * FFN];   // max K = 256 -> 8KB

    const int lane = threadIdx.x & 31;
    const int wave = threadIdx.x >> 5;
    const int tiles_n = Nc >> 4;
    const int total   = (M >> 4) * tiles_n;
    const int tile    = blockIdx.x * 8 + wave;
    const int tm      = (blockIdx.x * 8) / tiles_n;        // uniform per block

    // ---- stage A tile: 16 rows x K bf16 = 2*K x 16-byte chunks ----
    {
        const int cpr     = K >> 3;                        // 16B chunks per row
        const int nChunks = 16 * cpr;
        const char* Abase = (const char*)A32;
        for (int c = threadIdx.x; c < nChunks; c += blockDim.x) {
            const int row = c / cpr;
            const int off = (c - row * cpr) * 16;
            const char* gsrc = Abase + ((size_t)(tm * 16 + row) * K) * 2 + off;
            unsigned short* ldst = shA + ((size_t)row * K) + (off >> 1);
#if HAS_ASYNC_LDS
            __builtin_amdgcn_global_load_async_to_lds_b128(
                (gas_v4i*)(unsigned long long)(uintptr_t)gsrc,
                (las_v4i*)(unsigned)(uintptr_t)ldst, 0, 0);
#else
            *(uint4*)ldst = *(const uint4*)gsrc;
#endif
        }
#if HAS_ASYNC_LDS
#if __has_builtin(__builtin_amdgcn_s_wait_asynccnt)
        __builtin_amdgcn_s_wait_asynccnt(0);
#else
        asm volatile("s_wait_asynccnt 0x0" ::: "memory");
#endif
#endif
    }
    __syncthreads();

    if (tile < total) {
        const int tn   = tile % tiles_n;
        const int half = lane >> 4;                 // K-half select
        const int sub  = lane & 15;                 // row within tile
        const int Kd   = K >> 1;                    // dwords per row
        const unsigned int* aRow = (const unsigned int*)shA + (size_t)sub * Kd;
        const unsigned int* bRow = W32 + (size_t)(tn * 16 + sub) * Kd;

        v8f acc = {};
        for (int k0 = 0; k0 < K; k0 += 32) {
            union { v16bf v; unsigned int u[8]; } a, b;
            const int base = (k0 >> 1) + half * 4;
            #pragma unroll
            for (int q = 0; q < 4; ++q) {
                a.u[q]     = aRow[base + q];        // ds_load (LDS-staged A)
                a.u[4 + q] = aRow[base + 8 + q];
                b.u[q]     = bRow[base + q];        // global (L2-resident W)
                b.u[4 + q] = bRow[base + 8 + q];
            }
            acc = __builtin_amdgcn_wmma_f32_16x16x32_bf16(
                      false, a.v, false, b.v, (short)0, acc, false, false);
        }

        const int col = tn * 16 + sub;
        const float bv = (EPI == 2) ? 0.0f : bias[col];
        #pragma unroll
        for (int q = 0; q < 8; ++q) {
            int row = tm * 16 + half * 8 + q;       // CDNA5 f32 C/D layout
            float v = acc[q] + bv;
            if (EPI == 1) v = 0.5f * v * (1.0f + erff(v * 0.70710678118654752f));
            C[(size_t)row * Nc + col] = v;
        }
    }
}

// ---------------- edge scatter: one wave per edge, 8 cols per lane ----------
__global__ void scatter_edges(const int* __restrict__ src,
                              const int* __restrict__ dst,
                              const float* __restrict__ dinv,
                              const float* __restrict__ xw,   // [N,256]
                              float* __restrict__ acc,        // [N,256]
                              int E) {
    const int lane = threadIdx.x & 31;
    const int wave = threadIdx.x >> 5;
    const int e = blockIdx.x * 8 + wave;
    if (e >= E) return;
    const int s = src[e];
    const int d = dst[e];
    const float nrm = dinv[s] * dinv[d];
    const float4* xs = (const float4*)(xw + (size_t)s * FFN);
    float* ad = acc + (size_t)d * FFN;
    #pragma unroll
    for (int j = 0; j < 2; ++j) {
        float4 v = xs[lane * 2 + j];
        int c = (lane * 2 + j) * 4;
        gatomic_add(&ad[c + 0], v.x * nrm);
        gatomic_add(&ad[c + 1], v.y * nrm);
        gatomic_add(&ad[c + 2], v.z * nrm);
        gatomic_add(&ad[c + 3], v.w * nrm);
    }
}

// ---------------- gate combine: tanh(gate) * h -> bf16 ----------------------
__global__ void combine_gate(const float* __restrict__ gacc,
                             const float* __restrict__ xw,
                             const float* __restrict__ dinv,
                             const float* __restrict__ bgcn,
                             const float* __restrict__ h1,
                             unsigned short* __restrict__ h2,
                             size_t total) {
    size_t i = (size_t)blockIdx.x * blockDim.x + threadIdx.x;
    if (i >= total) return;
    const int node = (int)(i >> 8);
    const int c    = (int)(i & (FFN - 1));
    const float di = dinv[node];
    const float gate = gacc[i] + xw[i] * (2.0f * di * di) + bgcn[c];
    h2[i] = f2bf(tanhf(gate) * h1[i]);
}

// ---------------- host orchestration ----------------------------------------
extern "C" void kernel_launch(void* const* d_in, const int* in_sizes, int n_in,
                              void* d_out, int out_size, void* d_ws, size_t ws_size,
                              hipStream_t stream) {
    const float* x     = (const float*)d_in[0];
    const int*   ei    = (const int*)  d_in[1];
    const float* ln1_g = (const float*)d_in[2];
    const float* ln1_b = (const float*)d_in[3];
    const float* w_in  = (const float*)d_in[4];
    const float* b_in  = (const float*)d_in[5];
    const float* ln2_g = (const float*)d_in[6];
    const float* ln2_b = (const float*)d_in[7];
    const float* w_gcn = (const float*)d_in[8];
    const float* b_gcn = (const float*)d_in[9];
    const float* w_out = (const float*)d_in[10];
    const float* b_out = (const float*)d_in[11];
    float* out = (float*)d_out;

    const int N = in_sizes[0] / HID;
    const int E = in_sizes[1] / 2;
    const int* src = ei;
    const int* dst = ei + E;

    // workspace carving (256B aligned slots)
    char* ws = (char*)d_ws;
    size_t off = 0;
    auto carve = [&](size_t bytes) -> void* {
        void* p = ws + off;
        off = (off + bytes + 255) & ~(size_t)255;
        return p;
    };
    unsigned short* w_in_bf  = (unsigned short*)carve((size_t)FFN * HID * 2);
    unsigned short* w_gcn_bf = (unsigned short*)carve((size_t)FFN * FFN * 2);
    unsigned short* w_out_bf = (unsigned short*)carve((size_t)HID * FFN * 2);
    float*          deg      = (float*)carve((size_t)N * 4);
    float*          dinv     = (float*)carve((size_t)N * 4);
    unsigned short* a1       = (unsigned short*)carve((size_t)N * HID * 2);
    float*          h1       = (float*)carve((size_t)N * FFN * 4);
    unsigned short* gate_ln  = (unsigned short*)carve((size_t)N * FFN * 2);
    float*          xw       = (float*)carve((size_t)N * FFN * 4);
    float*          gacc     = (float*)carve((size_t)N * FFN * 4);
    unsigned short* h2       = (unsigned short*)carve((size_t)N * FFN * 2);
    (void)ws_size; (void)n_in; (void)out_size;

    const int B = 256;

    // weights -> bf16
    cast_f32_bf16<<<(FFN * HID + B - 1) / B, B, 0, stream>>>(w_in,  w_in_bf,  FFN * HID);
    cast_f32_bf16<<<(FFN * FFN + B - 1) / B, B, 0, stream>>>(w_gcn, w_gcn_bf, FFN * FFN);
    cast_f32_bf16<<<(HID * FFN + B - 1) / B, B, 0, stream>>>(w_out, w_out_bf, HID * FFN);

    // degree (self-loop weight 2.0 baked in) + accumulator init (every call!)
    fill_f32<<<(N + B - 1) / B, B, 0, stream>>>(deg, 2.0f, (size_t)N);
    fill_f32<<<((size_t)N * FFN + B - 1) / B, B, 0, stream>>>(gacc, 0.0f, (size_t)N * FFN);
    deg_accum<<<(E + B - 1) / B, B, 0, stream>>>(dst, deg, E);
    dinv_kernel<<<(N + B - 1) / B, B, 0, stream>>>(deg, dinv, N);

    // LN1 -> bf16 activations
    ln_rows<HID><<<(N + 7) / 8, B, 0, stream>>>(x, ln1_g, ln1_b, a1, N);

    // GEMM1 + GELU: h1 = gelu(a1 @ w_in^T + b_in)
    {
        int tiles = (N / 16) * (FFN / 16);
        gemm_wmma_bf16<1><<<(tiles + 7) / 8, B, 0, stream>>>(
            (const unsigned int*)a1, (const unsigned int*)w_in_bf, b_in, h1, N, HID, FFN);
    }

    // LN2 -> bf16 gate input
    ln_rows<FFN><<<(N + 7) / 8, B, 0, stream>>>(h1, ln2_g, ln2_b, gate_ln, N);

    // GEMM2 (no bias): xw = gate_ln @ w_gcn^T
    {
        int tiles = (N / 16) * (FFN / 16);
        gemm_wmma_bf16<2><<<(tiles + 7) / 8, B, 0, stream>>>(
            (const unsigned int*)gate_ln, (const unsigned int*)w_gcn_bf, nullptr, xw, N, FFN, FFN);
    }

    // edge aggregation: gacc[dst] += xw[src] * dinv[src]*dinv[dst]
    scatter_edges<<<(E + 7) / 8, B, 0, stream>>>(src, dst, dinv, xw, gacc, E);

    // gate = gacc + self-loop term + b_gcn ; h2 = tanh(gate) * h1  (bf16)
    {
        size_t total = (size_t)N * FFN;
        combine_gate<<<(total + B - 1) / B, B, 0, stream>>>(gacc, xw, dinv, b_gcn, h1, h2, total);
    }

    // GEMM3: out = h2 @ w_out^T + b_out
    {
        int tiles = (N / 16) * (HID / 16);
        gemm_wmma_bf16<0><<<(tiles + 7) / 8, B, 0, stream>>>(
            (const unsigned int*)h2, (const unsigned int*)w_out_bf, b_out, out, N, FFN, HID);
    }
}
